// RegionProposalNetwork_59751585022497
// MI455X (gfx1250) — matrix operand order repaired and checked by reading, not compiled, not run on hardware
//
#include <hip/hip_runtime.h>
#include <hip/hip_bf16.h>

typedef __attribute__((ext_vector_type(16))) _Float16 v16h;
typedef __attribute__((ext_vector_type(8)))  _Float16 v8h;
typedef __attribute__((ext_vector_type(8)))  float    v8f;
typedef __attribute__((ext_vector_type(4)))  float    v4f;

#define BATCH   4
#define CIN     512
#define CMID    512
#define HH      64
#define WW      64
#define HW      4096            // H*W
#define NTOT    16384           // B*H*W
#define K1      4608            // CIN*9
#define NANCH   9
#define NBOX    36864           // HW*NANCH
#define NPOST   300

union AFrag { v16h v; v8h h[2]; };

// ---------------------------------------------------------------------------
// Kernel 1: 3x3 conv as implicit GEMM  [512 x 4608] x [4608 x 16384]
// workgroup tile 128(M) x 128(N), 8 waves (4M x 2N), each wave 32x64:
// acc[2][4] -> 8 WMMAs per 32-K step. Incremental k -> (c, r, s) decomposition
// keeps the im2col address math division-free inside the loop.
// ---------------------------------------------------------------------------
__global__ __launch_bounds__(256) void conv3_wmma(
    const float* __restrict__ x, const float* __restrict__ w,
    const float* __restrict__ bias, _Float16* __restrict__ mid)
{
    __shared__ __align__(64) _Float16 As[128][32];   // [m][k]
    __shared__ __align__(64) _Float16 Bs[128][32];   // [n][k] (fragment-ready)

    const int tid  = threadIdx.x;
    const int lane = tid & 31;
    const int wave = tid >> 5;          // 0..7
    const int wm   = wave >> 1;         // 0..3  -> m offset wm*32
    const int wn   = wave & 1;          // 0..1  -> n offset wn*64
    const int m0   = blockIdx.y * 128;
    const int n0   = blockIdx.x * 128;

    // block's 128 n values: one batch image, rows yRow and yRow+1
    const int bImg = n0 >> 12;
    const int yRow = (n0 >> 6) & 63;

    v8f acc[2][4];
    {
        v8f z = {};
#pragma unroll
        for (int i = 0; i < 2; ++i)
#pragma unroll
            for (int j = 0; j < 4; ++j) acc[i][j] = z;
    }

    // A staging: thread t loads 16 contiguous k for row am
    const int am = tid >> 1;            // 0..127
    const int ak = (tid & 1) * 16;      // 0 / 16
    // B staging: thread t owns k-row bkl and 16 consecutive n
    const int bkl = tid >> 3;           // 0..31
    const int nl0 = (tid & 7) * 16;     // 0..112
    const int ys  = nl0 >> 6;           // sub-row 0/1
    const int x0  = nl0 & 63;           // 0,16,32,48

    // one-time decomposition of this thread's staging k; updated incrementally
    int c  = bkl / 9;
    int rs = bkl - c * 9;

    const float* xb = x + (size_t)bImg * (CIN * HW);

    for (int k0 = 0; k0 < K1; k0 += 32) {
        __syncthreads();
        // ---- stage A (weights already row-major [m][k]) ----
        {
            const float* src = w + (size_t)(m0 + am) * K1 + k0 + ak;
            if (k0 + 32 < K1) __builtin_prefetch(src + 32, 0, 1);
            const v4f f0 = *(const v4f*)(src);
            const v4f f1 = *(const v4f*)(src + 4);
            const v4f f2 = *(const v4f*)(src + 8);
            const v4f f3 = *(const v4f*)(src + 12);
            v8h h0, h1;
#pragma unroll
            for (int j = 0; j < 4; ++j) {
                h0[j]     = (_Float16)f0[j];
                h0[4 + j] = (_Float16)f1[j];
                h1[j]     = (_Float16)f2[j];
                h1[4 + j] = (_Float16)f3[j];
            }
            *(v8h*)&As[am][ak]     = h0;
            *(v8h*)&As[am][ak + 8] = h1;
        }
        // ---- stage B (im2col on the fly, division-free) ----
        {
            const int r  = (rs >= 6) ? 2 : ((rs >= 3) ? 1 : 0);
            const int dx = rs - r * 3 - 1;
            const int yy = yRow + ys + r - 1;
            const bool yok = (yy >= 0) && (yy < HH);
            const float* src = xb + (size_t)c * HW + yy * WW;
#pragma unroll
            for (int j = 0; j < 16; ++j) {
                const int xx = x0 + j + dx;
                float v = (yok && xx >= 0 && xx < WW) ? src[xx] : 0.0f;
                Bs[nl0 + j][bkl] = (_Float16)v;
            }
            // advance k by 32: 32 = 3*9 + 5
            c += 3; rs += 5;
            if (rs >= 9) { rs -= 9; ++c; }
        }
        __syncthreads();

        // ---- fragments: contiguous k = 16*(lane/16) + s, same order for A and B ----
        const int fl  = lane & 15;
        const int kbA = (lane >> 4) * 16;    // A: lanes 0-15 K=0..15, 16-31 K=16..31
        const int kbB = (lane >> 4) * 16;    // B: lanes 0-15 K=0..15, 16-31 K=16..31
        v16h afr[2], bfr[4];
#pragma unroll
        for (int s = 0; s < 2; ++s) {
            const _Float16* ap = &As[wm * 32 + s * 16 + fl][0];
            AFrag af;
            af.h[0] = *(const v8h*)(ap + kbA);
            af.h[1] = *(const v8h*)(ap + kbA + 8);
            afr[s] = af.v;
        }
#pragma unroll
        for (int j = 0; j < 4; ++j)
            bfr[j] = *(const v16h*)(&Bs[wn * 64 + j * 16 + fl][kbB]);
#pragma unroll
        for (int i = 0; i < 2; ++i)
#pragma unroll
            for (int j = 0; j < 4; ++j)
                acc[i][j] = __builtin_amdgcn_wmma_f32_16x16x32_f16(
                    false, afr[i], false, bfr[j], (short)0, acc[i][j], false, false);
    }

    // ---- epilogue: bias + ReLU, store f16 mid[b][m][p] ----
    const int nl   = lane & 15;
    const int mrow = (lane >> 4) * 8;
#pragma unroll
    for (int i = 0; i < 2; ++i) {
#pragma unroll
        for (int j = 0; j < 4; ++j) {
            const int n = n0 + wn * 64 + j * 16 + nl;
            const int p = n & 4095;
#pragma unroll
            for (int v = 0; v < 8; ++v) {
                const int m = m0 + wm * 32 + i * 16 + mrow + v;
                float val = acc[i][j][v] + bias[m];
                val = val > 0.0f ? val : 0.0f;
                mid[(size_t)bImg * (CMID * HW) + (size_t)m * HW + p] = (_Float16)val;
            }
        }
    }
}

// ---------------------------------------------------------------------------
// Kernel 2: fused 1x1 loc(36) + cls(18) heads, GEMM M=64(pad) x K=512 x N=16384
// workgroup tile 64x64, 8 waves (4 in M x 2 in N), each wave 16x32
// ---------------------------------------------------------------------------
__global__ __launch_bounds__(256) void head_wmma(
    const _Float16* __restrict__ mid,
    const float* __restrict__ locw, const float* __restrict__ locb,
    const float* __restrict__ clsw, const float* __restrict__ clsb,
    float* __restrict__ outScores, float* __restrict__ outLocs)
{
    __shared__ __align__(64) _Float16 As[64][32];
    __shared__ __align__(64) _Float16 Bs[64][32];

    const int tid  = threadIdx.x;
    const int lane = tid & 31;
    const int wave = tid >> 5;
    const int wm   = wave & 3;          // m tile = wm*16
    const int wn   = wave >> 2;         // n tile = wn*32
    const int n0   = blockIdx.x * 64;
    const int bImg = n0 >> 12;
    const int yRow = (n0 >> 6) & 63;

    v8f acc[2];
    { v8f z = {}; acc[0] = z; acc[1] = z; }

    const int am  = tid >> 2;           // 0..63
    const int ak  = (tid & 3) * 8;      // 0,8,16,24
    const int bk  = tid >> 3;           // 0..31
    const int bx0 = (tid & 7) * 8;

    for (int k0 = 0; k0 < CMID; k0 += 32) {
        __syncthreads();
        // A = concatenated [loc_w(36); cls_w(18); zeros(10)] rows
        {
#pragma unroll
            for (int j = 0; j < 8; ++j) {
                const int kk = k0 + ak + j;
                float v = 0.0f;
                if (am < 36)       v = locw[(size_t)am * CMID + kk];
                else if (am < 54)  v = clsw[(size_t)(am - 36) * CMID + kk];
                As[am][ak + j] = (_Float16)v;
            }
        }
        // B = mid activations (already f16)
        {
            const _Float16* src = mid + ((size_t)bImg * CMID + k0 + bk) * HW + yRow * WW + bx0;
#pragma unroll
            for (int j = 0; j < 8; ++j) Bs[bx0 + j][bk] = src[j];
        }
        __syncthreads();

        const int fl  = lane & 15;
        const int kbA = (lane >> 4) * 16;
        const int kbB = (lane >> 4) * 16;
        AFrag af;
        {
            const _Float16* ap = &As[wm * 16 + fl][0];
            af.h[0] = *(const v8h*)(ap + kbA);
            af.h[1] = *(const v8h*)(ap + kbA + 8);
        }
#pragma unroll
        for (int s = 0; s < 2; ++s) {
            v16h bfr = *(const v16h*)(&Bs[wn * 32 + s * 16 + fl][kbB]);
            acc[s] = __builtin_amdgcn_wmma_f32_16x16x32_f16(
                false, af.v, false, bfr, (short)0, acc[s], false, false);
        }
    }

    // epilogue: scatter into transposed (NHWC) loc/score layouts with bias
    const int nl   = lane & 15;
    const int mrow = (lane >> 4) * 8;
#pragma unroll
    for (int s = 0; s < 2; ++s) {
        const int n = n0 + wn * 32 + s * 16 + nl;
        const int b = n >> 12;
        const int p = n & 4095;
#pragma unroll
        for (int v = 0; v < 8; ++v) {
            const int m = wm * 16 + mrow + v;
            const float val = acc[s][v];
            if (m < 36) {
                outLocs[(size_t)b * (NBOX * 4) + (size_t)p * 36 + m] = val + locb[m];
            } else if (m < 54) {
                const int cm = m - 36;
                outScores[(size_t)b * (NBOX * 2) + (size_t)p * 18 + cm] = val + clsb[cm];
            }
        }
    }
}

// ---------------------------------------------------------------------------
// Kernel 3: softmax-fg + box decode + clip + min-size masking
// ---------------------------------------------------------------------------
__global__ void decode_kernel(
    const float* __restrict__ scores, const float* __restrict__ locs,
    const float* __restrict__ anchor, const float* __restrict__ imsize,
    float* __restrict__ boxes, float* __restrict__ sc)
{
    const int gid = blockIdx.x * blockDim.x + threadIdx.x;
    if (gid >= BATCH * NBOX) return;
    const int b = gid / NBOX;
    const int i = gid - b * NBOX;

    const float* l = locs + (size_t)b * (NBOX * 4) + (size_t)i * 4;
    const float s0 = scores[(size_t)b * (NBOX * 2) + (size_t)i * 2 + 0];
    const float s1 = scores[(size_t)b * (NBOX * 2) + (size_t)i * 2 + 1];
    const float fg = 1.0f / (1.0f + __expf(s0 - s1));

    const float* a = anchor + (size_t)i * 4;
    const float ah = a[2] - a[0], aw = a[3] - a[1];
    const float acy = a[0] + 0.5f * ah, acx = a[1] + 0.5f * aw;
    const float cy = l[0] * ah + acy, cx = l[1] * aw + acx;
    const float h = __expf(l[2]) * ah, w = __expf(l[3]) * aw;
    const float Hs = imsize[0], Ws = imsize[1];
    float y0 = cy - 0.5f * h, x0 = cx - 0.5f * w;
    float y1 = cy + 0.5f * h, x1 = cx + 0.5f * w;
    y0 = fminf(fmaxf(y0, 0.0f), Hs);  x0 = fminf(fmaxf(x0, 0.0f), Ws);
    y1 = fminf(fmaxf(y1, 0.0f), Hs);  x1 = fminf(fmaxf(x1, 0.0f), Ws);

    float* bo = boxes + (size_t)gid * 4;
    bo[0] = y0; bo[1] = x0; bo[2] = y1; bo[3] = x1;
    const bool valid = ((y1 - y0) >= 16.0f) && ((x1 - x0) >= 16.0f);
    sc[gid] = valid ? fg : -__builtin_inff();
}

// ---------------------------------------------------------------------------
// Kernel 4: greedy NMS, one block per batch image, 300 picks
// ---------------------------------------------------------------------------
#define NMS_T 256
__global__ __launch_bounds__(NMS_T) void nms_kernel(
    const float* __restrict__ boxes, const float* __restrict__ sc,
    float* __restrict__ rois)
{
    __shared__ unsigned char supp[NBOX];
    __shared__ float rs[NMS_T];
    __shared__ int   ri[NMS_T];
    __shared__ float bb[5];   // best box + ok flag

    const int b = blockIdx.x;
    const int t = threadIdx.x;
    const float* BX = boxes + (size_t)b * NBOX * 4;
    const float* S  = sc    + (size_t)b * NBOX;
    float* out = rois + (size_t)b * NPOST * 4;

    for (int i = t; i < NBOX; i += NMS_T) supp[i] = 0;
    __syncthreads();

    for (int it = 0; it < NPOST; ++it) {
        // local argmax (strided -> coalesced; ascending keeps lowest index on ties)
        float best = -__builtin_inff();
        int   bi   = -1;
        for (int i = t; i < NBOX; i += NMS_T) {
            if (!supp[i]) {
                const float v = S[i];
                if (v > best) { best = v; bi = i; }
            }
        }
        rs[t] = best; ri[t] = bi;
        __syncthreads();
        for (int off = NMS_T >> 1; off > 0; off >>= 1) {
            if (t < off) {
                const float so = rs[t + off];
                const int   io = ri[t + off];
                if (so > rs[t] || (so == rs[t] && (unsigned)io < (unsigned)ri[t])) {
                    rs[t] = so; ri[t] = io;
                }
            }
            __syncthreads();
        }
        if (t == 0) {
            const int   idx = ri[0];
            const bool  ok  = (idx >= 0) && (rs[0] > -__builtin_inff());
            float b0 = 0.f, b1 = 0.f, b2 = 0.f, b3 = 0.f;
            if (ok) {
                b0 = BX[idx * 4 + 0]; b1 = BX[idx * 4 + 1];
                b2 = BX[idx * 4 + 2]; b3 = BX[idx * 4 + 3];
                supp[idx] = 1;
            }
            bb[0] = b0; bb[1] = b1; bb[2] = b2; bb[3] = b3;
            bb[4] = ok ? 1.0f : 0.0f;
            out[it * 4 + 0] = b0; out[it * 4 + 1] = b1;
            out[it * 4 + 2] = b2; out[it * 4 + 3] = b3;
        }
        __syncthreads();
        if (bb[4] > 0.0f) {
            const float by0 = bb[0], bx0 = bb[1], by1 = bb[2], bx1 = bb[3];
            const float areaB = (by1 - by0) * (bx1 - bx0);
            for (int i = t; i < NBOX; i += NMS_T) {
                const float cy0 = BX[i * 4 + 0], cx0 = BX[i * 4 + 1];
                const float cy1 = BX[i * 4 + 2], cx1 = BX[i * 4 + 3];
                const float iy = fmaxf(fminf(by1, cy1) - fmaxf(by0, cy0), 0.0f);
                const float ix = fmaxf(fminf(bx1, cx1) - fmaxf(bx0, cx0), 0.0f);
                const float inter = iy * ix;
                const float areaC = (cy1 - cy0) * (cx1 - cx0);
                const float iou = inter / (areaB + areaC - inter + 1e-9f);
                if (iou > 0.7f) supp[i] = 1;
            }
        }
        __syncthreads();
    }
}

// ---------------------------------------------------------------------------
// Kernel 5: roi_indices = repeat(arange(B), NPOST)
// ---------------------------------------------------------------------------
__global__ void fill_indices(float* __restrict__ outIdx)
{
    const int i = blockIdx.x * blockDim.x + threadIdx.x;
    if (i < BATCH * NPOST) outIdx[i] = (float)(i / NPOST);
}

// ---------------------------------------------------------------------------
extern "C" void kernel_launch(void* const* d_in, const int* in_sizes, int n_in,
                              void* d_out, int out_size, void* d_ws, size_t ws_size,
                              hipStream_t stream) {
    const float* x       = (const float*)d_in[0];
    const float* anchor  = (const float*)d_in[1];
    const float* imsize  = (const float*)d_in[2];
    const float* conv_w  = (const float*)d_in[3];
    const float* conv_b  = (const float*)d_in[4];
    const float* loc_w   = (const float*)d_in[5];
    const float* loc_b   = (const float*)d_in[6];
    const float* cls_w   = (const float*)d_in[7];
    const float* cls_b   = (const float*)d_in[8];

    float* out = (float*)d_out;
    float* outScores = out;                      // 4*36864*2 = 294912
    float* outLocs   = out + 294912;             // 4*36864*4 = 589824
    float* outRois   = out + 884736;             // 1200*4    = 4800
    float* outIdx    = out + 889536;             // 1200

    _Float16* mid = (_Float16*)d_ws;                             // 16 MiB f16
    float* boxes  = (float*)((char*)d_ws + (size_t)16777216);    // 4*36864*4 f32
    float* scmask = boxes + (size_t)BATCH * NBOX * 4;            // 4*36864 f32

    conv3_wmma<<<dim3(NTOT / 128, CMID / 128), 256, 0, stream>>>(x, conv_w, conv_b, mid);
    head_wmma<<<dim3(NTOT / 64), 256, 0, stream>>>(mid, loc_w, loc_b, cls_w, cls_b,
                                                   outScores, outLocs);
    decode_kernel<<<(BATCH * NBOX + 255) / 256, 256, 0, stream>>>(
        outScores, outLocs, anchor, imsize, boxes, scmask);
    nms_kernel<<<BATCH, NMS_T, 0, stream>>>(boxes, scmask, outRois);
    fill_indices<<<(BATCH * NPOST + 255) / 256, 256, 0, stream>>>(outIdx);
}
